// RSSMCore_17257178595401
// MI455X (gfx1250) — compile-verified, matrix operand order checked by reading, not run on hardware
//
#include <hip/hip_runtime.h>
#include <hip/hip_bf16.h>
#include <math.h>

typedef __attribute__((ext_vector_type(16))) __bf16 v16bf;
typedef __attribute__((ext_vector_type(8)))  float  v8f;
typedef __attribute__((ext_vector_type(4)))  int    v4i;

union FragA { v16bf v; unsigned int u[8]; };

#if defined(__gfx1250__) && __has_builtin(__builtin_amdgcn_global_load_async_to_lds_b128)
#define HAVE_ASYNC_LDS 1
#else
#define HAVE_ASYNC_LDS 0
#endif

static __device__ __forceinline__ void wait_async0() {
#if HAVE_ASYNC_LDS
#if __has_builtin(__builtin_amdgcn_s_wait_asynccnt)
    __builtin_amdgcn_s_wait_asynccnt(0);
#else
    asm volatile("s_wait_asynccnt 0x0" ::: "memory");
#endif
#endif
}

// ---------------------------------------------------------------------------
// Weight pre-pass: out[n*K + k] = (bf16) in[k*N + n]   (transpose + convert)
// ---------------------------------------------------------------------------
__global__ __launch_bounds__(256) void transpose_to_bf16(
    const float* __restrict__ in, __bf16* __restrict__ out, int K, int N)
{
    int i = blockIdx.x * 256 + threadIdx.x;   // < K*N (fits in int)
    int n = i / K;
    int k = i - n * K;
    out[i] = (__bf16)in[(size_t)k * N + n];
}

// ---------------------------------------------------------------------------
// Tiled bf16 WMMA GEMM:  Cout[M,N] = A[M,K](fp32,lda) * BT[N,K](bf16) (+bias)(+Cin)
// Software-pipelined, double-buffered LDS. B tiles prefetched via
// GLOBAL_LOAD_ASYNC_TO_LDS_B128 (ASYNCcnt); A tiles prefetched into registers,
// fp32->bf16 converted, stored after WMMA issue. Block 128x128x32, 8 wave32,
// wave grid 2x4, 4x2 WMMA tiles per wave.
// ---------------------------------------------------------------------------
__global__ __launch_bounds__(256) void gemm_bf16_wmma(
    const float* __restrict__ A, int lda,
    const __bf16* __restrict__ BT,          // [N][K] bf16, ldb == K
    const float* __restrict__ bias,         // may be null
    const float* __restrict__ Cin,          // may be null (ld = N)
    float* __restrict__ Cout,               // ld = N
    int M, int N, int K)
{
    constexpr int BM = 128, BN = 128, BK = 32, LDK = BK + 8;  // stride 40
    __shared__ __align__(16) __bf16 As[2][BM][LDK];
    __shared__ __align__(16) __bf16 Bs[2][BN][LDK];

    const int tid  = threadIdx.x;
    const int lane = tid & 31;
    const int wave = tid >> 5;          // 0..7
    const int wm   = wave >> 2;         // 0..1  -> 64 rows each
    const int wn   = wave & 3;          // 0..3  -> 32 cols each
    const int half = lane >> 4;         // 0/1
    const int l16  = lane & 15;
    const int bM   = blockIdx.y * BM;
    const int bN   = blockIdx.x * BN;

    v8f acc[4][2];
    for (int i = 0; i < 4; ++i)
        for (int j = 0; j < 2; ++j)
            for (int v = 0; v < 8; ++v) acc[i][j][v] = 0.0f;

    float2 areg[8];

    auto issueB = [&](int k0, int s) {
        #pragma unroll
        for (int i = 0; i < 2; ++i) {
            int idx = tid + i * 256;          // 0..511 octets of bf16
            int n   = idx >> 2;
            int kg  = (idx & 3) << 3;
            const __bf16* src = BT + (size_t)(bN + n) * K + k0 + kg;
#if HAVE_ASYNC_LDS
            __builtin_amdgcn_global_load_async_to_lds_b128(
                (__attribute__((address_space(1))) v4i*)src,
                (__attribute__((address_space(3))) v4i*)&Bs[s][n][kg], 0, 0);
#else
            *(uint4*)&Bs[s][n][kg] = *(const uint4*)src;
#endif
        }
    };
    auto loadA = [&](int k0) {
        #pragma unroll
        for (int i = 0; i < 8; ++i) {
            int idx = tid + i * 256;          // pair index 0..2047
            int m   = idx >> 4;               // 16 pairs per row
            int kp  = (idx & 15) << 1;
            areg[i] = *(const float2*)(A + (size_t)(bM + m) * lda + k0 + kp);
        }
    };
    auto storeA = [&](int s) {
        #pragma unroll
        for (int i = 0; i < 8; ++i) {
            int idx = tid + i * 256;
            int m   = idx >> 4;
            int kp  = (idx & 15) << 1;
            union { __bf16 h[2]; unsigned int u; } pk;
            pk.h[0] = (__bf16)areg[i].x;
            pk.h[1] = (__bf16)areg[i].y;
            *(unsigned int*)&As[s][m][kp] = pk.u;
        }
    };

    // prologue: stage k-block 0 into buffer 0
    issueB(0, 0);
    loadA(0);
    storeA(0);

    const int nst = K / BK;
    for (int it = 0; it < nst; ++it) {
        const int cur = it & 1;
        wait_async0();          // async B tile for 'cur' has landed
        __syncthreads();        // publish As[cur]/Bs[cur]; retire reads of cur^1

        const bool more = (it + 1 < nst);
        if (more) {
            issueB((it + 1) * BK, cur ^ 1);   // DMA next B tile (ASYNCcnt)
            loadA((it + 1) * BK);             // prefetch next A tile into regs
        }

        FragA afrag[4], bfrag[2];
        #pragma unroll
        for (int tm = 0; tm < 4; ++tm) {
            int row = wm * 64 + tm * 16 + l16;
            #pragma unroll
            for (int p = 0; p < 8; ++p) {
                int kb = ((p < 4) ? 2 * p : 16 + 2 * (p - 4)) + 8 * half;
                afrag[tm].u[p] = *(const unsigned int*)&As[cur][row][kb];
            }
        }
        #pragma unroll
        for (int tn = 0; tn < 2; ++tn) {
            int col = wn * 32 + tn * 16 + l16;
            #pragma unroll
            for (int p = 0; p < 8; ++p) {
                int kb = ((p < 4) ? 2 * p : 16 + 2 * (p - 4)) + 8 * half;
                bfrag[tn].u[p] = *(const unsigned int*)&Bs[cur][col][kb];
            }
        }
        #pragma unroll
        for (int tm = 0; tm < 4; ++tm)
            #pragma unroll
            for (int tn = 0; tn < 2; ++tn)
                acc[tm][tn] = __builtin_amdgcn_wmma_f32_16x16x32_bf16(
                    false, afrag[tm].v, false, bfrag[tn].v,
                    (short)0, acc[tm][tn], false, false);

        if (more) storeA(cur ^ 1);  // convert + store next A tile (overlaps WMMA)
    }

    // epilogue: C layout = VGPR v -> M = v + 8*half ; lane%16 -> N
    #pragma unroll
    for (int tm = 0; tm < 4; ++tm) {
        #pragma unroll
        for (int tn = 0; tn < 2; ++tn) {
            int n = bN + wn * 32 + tn * 16 + l16;
            float bv = bias ? bias[n] : 0.0f;
            #pragma unroll
            for (int v = 0; v < 8; ++v) {
                int m = bM + wm * 64 + tm * 16 + v + 8 * half;
                size_t off = (size_t)m * N + n;
                float val = acc[tm][tn][v] + bv;
                if (Cin) val += Cin[off];
                Cout[off] = val;
            }
        }
    }
}

// ---------------------------------------------------------------------------
// LayerNorm + ELU over rows of width W (in-place). One block per row.
// ---------------------------------------------------------------------------
__global__ __launch_bounds__(256) void ln_elu_kernel(
    float* __restrict__ X, const float* __restrict__ g,
    const float* __restrict__ b, int W)
{
    __shared__ float red[256];
    float* x = X + (size_t)blockIdx.x * W;

    float s = 0.0f;
    for (int i = threadIdx.x; i < W; i += 256) s += x[i];
    red[threadIdx.x] = s; __syncthreads();
    for (int o = 128; o > 0; o >>= 1) {
        if (threadIdx.x < o) red[threadIdx.x] += red[threadIdx.x + o];
        __syncthreads();
    }
    float mean = red[0] / (float)W;
    __syncthreads();

    float vs = 0.0f;
    for (int i = threadIdx.x; i < W; i += 256) { float d = x[i] - mean; vs += d * d; }
    red[threadIdx.x] = vs; __syncthreads();
    for (int o = 128; o > 0; o >>= 1) {
        if (threadIdx.x < o) red[threadIdx.x] += red[threadIdx.x + o];
        __syncthreads();
    }
    float inv = rsqrtf(red[0] / (float)W + 1e-5f);

    for (int i = threadIdx.x; i < W; i += 256) {
        float y = (x[i] - mean) * inv * g[i] + b[i];
        x[i] = (y > 0.0f) ? y : expm1f(y);
    }
}

// dst[b*W + w] = src[b*W + w] * (resets[b] ? 0 : 1)
__global__ __launch_bounds__(256) void mask_kernel(
    const float* __restrict__ src, float* __restrict__ dst,
    const unsigned char* __restrict__ resets_t, int W, int n)
{
    int i = blockIdx.x * 256 + threadIdx.x;
    if (i >= n) return;
    int b = i / W;
    dst[i] = resets_t[b] ? 0.0f : src[i];
}

// GRU gate combine (gates order r,z,n). Writes new h into features slice + h_cur.
__global__ __launch_bounds__(256) void gru_kernel(
    const float* __restrict__ gi, const float* __restrict__ gh,
    const float* __restrict__ hprev, float* __restrict__ h_feat,
    float* __restrict__ h_cur, int D, int FS, int n)
{
    int i = blockIdx.x * 256 + threadIdx.x;
    if (i >= n) return;
    int b = i / D, d = i - b * D;
    size_t base = (size_t)b * 3 * D;
    float ir = gi[base + d], iz = gi[base + D + d], in_ = gi[base + 2 * D + d];
    float hr = gh[base + d], hz = gh[base + D + d], hn  = gh[base + 2 * D + d];
    float r  = 1.0f / (1.0f + expf(-(ir + hr)));
    float z  = 1.0f / (1.0f + expf(-(iz + hz)));
    float nn = tanhf(in_ + r * hn);
    float h  = (1.0f - z) * nn + z * hprev[i];
    h_cur[i] = h;
    h_feat[(size_t)b * FS + d] = h;
}

// sample = mean + (softplus(std_raw)+0.1)*noise ; writes features slice + z_cur
__global__ __launch_bounds__(256) void sample_kernel(
    const float* __restrict__ post, const float* __restrict__ noise_t,
    float* __restrict__ feat, float* __restrict__ z_cur,
    int S, int D, int FS, int n)
{
    int i = blockIdx.x * 256 + threadIdx.x;
    if (i >= n) return;
    int b = i / S, s = i - b * S;
    float mean = post[(size_t)b * 2 * S + s];
    float sr   = post[(size_t)b * 2 * S + S + s];
    float sp   = (sr > 30.0f) ? sr : log1pf(expf(sr));
    float smp  = mean + (sp + 0.1f) * noise_t[i];
    z_cur[i] = smp;
    feat[(size_t)b * FS + D + s] = smp;
}

// ---------------------------------------------------------------------------
extern "C" void kernel_launch(void* const* d_in, const int* in_sizes, int n_in,
                              void* d_out, int out_size, void* d_ws, size_t ws_size,
                              hipStream_t stream)
{
    constexpr int T = 64, B = 256, E = 1536, AA = 32, D = 2048, H = 1024, S = 1024;
    constexpr int FS = D + S;   // features row stride 3072

    const float* embeds  = (const float*)d_in[0];
    const float* actions = (const float*)d_in[1];
    const unsigned char* resets = (const unsigned char*)d_in[2];  // bool [T,B,1]
    const float* in_h = (const float*)d_in[3];
    const float* in_z = (const float*)d_in[4];
    const float* noise = (const float*)d_in[5];
    const float* Wz  = (const float*)d_in[6];
    const float* bz  = (const float*)d_in[7];
    const float* Wa  = (const float*)d_in[8];
    const float* g_in = (const float*)d_in[9];
    const float* b_in = (const float*)d_in[10];
    const float* Wih = (const float*)d_in[11];
    const float* Whh = (const float*)d_in[12];
    const float* bih = (const float*)d_in[13];
    const float* bhh = (const float*)d_in[14];
    const float* Wph = (const float*)d_in[15];
    const float* bph = (const float*)d_in[16];
    const float* Wpe = (const float*)d_in[17];
    const float* g_po = (const float*)d_in[18];
    const float* b_po = (const float*)d_in[19];
    const float* Wp  = (const float*)d_in[20];
    const float* bp  = (const float*)d_in[21];
    const float* Wprh = (const float*)d_in[22];
    const float* bprh = (const float*)d_in[23];
    const float* g_pr = (const float*)d_in[24];
    const float* b_pr = (const float*)d_in[25];
    const float* Wpr = (const float*)d_in[26];
    const float* bpr = (const float*)d_in[27];

    // output layout (flat, return order)
    float* out     = (float*)d_out;
    float* priors  = out;
    float* posts   = priors + (size_t)T * B * 2 * S;
    float* feats   = posts  + (size_t)T * B * 2 * S;
    float* hf      = feats  + (size_t)T * B * FS;
    float* zf      = hf     + (size_t)B * D;

    // workspace: bf16 transposed weights first, then fp32 scratch
    char* wsb = (char*)d_ws;
    auto carve_bf = [&](size_t elems) { __bf16* p = (__bf16*)wsb; wsb += elems * 2; return p; };
    __bf16* WzT   = carve_bf((size_t)H * S);        // [H][S]
    __bf16* WaT   = carve_bf((size_t)H * AA);       // [H][A]
    __bf16* WihT  = carve_bf((size_t)3 * D * H);    // [3D][H]
    __bf16* WhhT  = carve_bf((size_t)3 * D * D);    // [3D][D]
    __bf16* WphT  = carve_bf((size_t)H * D);        // [H][D]
    __bf16* WpeT  = carve_bf((size_t)H * E);        // [H][E]
    __bf16* WpT   = carve_bf((size_t)2 * S * H);    // [2S][H]
    __bf16* WprhT = carve_bf((size_t)H * D);        // [H][D]
    __bf16* WprT  = carve_bf((size_t)2 * S * H);    // [2S][H]

    float* fws   = (float*)wsb;
    float* hm    = fws;                   // B*D masked h
    float* zm    = hm + (size_t)B * D;    // B*S masked z
    float* xbuf  = zm + (size_t)B * S;    // B*H
    float* gi    = xbuf + (size_t)B * H;  // B*3D
    float* gh    = gi + (size_t)B * 3 * D;// B*3D
    float* pxbuf = gh + (size_t)B * 3 * D;// B*H
    float* h_cur = pxbuf + (size_t)B * H; // B*D
    float* z_cur = h_cur + (size_t)B * D; // B*S

    // one-time (per launch) weight transpose + bf16 convert
    auto conv = [&](const float* w, __bf16* wt, int K, int N) {
        transpose_to_bf16<<<(K * N) / 256, 256, 0, stream>>>(w, wt, K, N);
    };
    conv(Wz,   WzT,   S,  H);
    conv(Wa,   WaT,   AA, H);
    conv(Wih,  WihT,  H,  3 * D);
    conv(Whh,  WhhT,  D,  3 * D);
    conv(Wph,  WphT,  D,  H);
    conv(Wpe,  WpeT,  E,  H);
    conv(Wp,   WpT,   H,  2 * S);
    conv(Wprh, WprhT, D,  H);
    conv(Wpr,  WprT,  H,  2 * S);

    (void)hipMemcpyAsync(h_cur, in_h, (size_t)B * D * sizeof(float),
                         hipMemcpyDeviceToDevice, stream);
    (void)hipMemcpyAsync(z_cur, in_z, (size_t)B * S * sizeof(float),
                         hipMemcpyDeviceToDevice, stream);

    auto gemm = [&](const float* A, int lda, const __bf16* BT,
                    const float* bias, const float* Cin, float* Cout,
                    int M, int N, int K) {
        dim3 grid(N / 128, M / 128);
        gemm_bf16_wmma<<<grid, 256, 0, stream>>>(A, lda, BT, bias, Cin,
                                                 Cout, M, N, K);
    };

    for (int t = 0; t < T; ++t) {
        const unsigned char* rs = resets + (size_t)t * B;
        mask_kernel<<<(B * D) / 256, 256, 0, stream>>>(h_cur, hm, rs, D, B * D);
        mask_kernel<<<(B * S) / 256, 256, 0, stream>>>(z_cur, zm, rs, S, B * S);

        // x = elu(ln(zm@Wz + bz + a_t@Wa))
        gemm(zm, S, WzT, bz, nullptr, xbuf, B, H, S);
        gemm(actions + (size_t)t * B * AA, AA, WaT, nullptr, xbuf, xbuf, B, H, AA);
        ln_elu_kernel<<<B, 256, 0, stream>>>(xbuf, g_in, b_in, H);

        // GRU gates
        gemm(xbuf, H, WihT, bih, nullptr, gi, B, 3 * D, H);
        gemm(hm,   D, WhhT, bhh, nullptr, gh, B, 3 * D, D);
        gru_kernel<<<(B * D) / 256, 256, 0, stream>>>(
            gi, gh, hm, feats + (size_t)t * B * FS, h_cur, D, FS, B * D);

        // px = elu(ln(h@Wph + bph + e_t@Wpe))
        gemm(h_cur, D, WphT, bph, nullptr, pxbuf, B, H, D);
        gemm(embeds + (size_t)t * B * E, E, WpeT, nullptr, pxbuf, pxbuf, B, H, E);
        ln_elu_kernel<<<B, 256, 0, stream>>>(pxbuf, g_po, b_po, H);

        // post = px@Wp + bp  (directly into d_out posts slice)
        gemm(pxbuf, H, WpT, bp, nullptr, posts + (size_t)t * B * 2 * S,
             B, 2 * S, H);
        sample_kernel<<<(B * S) / 256, 256, 0, stream>>>(
            posts + (size_t)t * B * 2 * S, noise + (size_t)t * B * S,
            feats + (size_t)t * B * FS, z_cur, S, D, FS, B * S);
    }

    // batch prior over states_h stored in the features slices (lda = FS)
    for (int t = 0; t < T; ++t) {
        gemm(feats + (size_t)t * B * FS, FS, WprhT, bprh, nullptr, pxbuf,
             B, H, D);
        ln_elu_kernel<<<B, 256, 0, stream>>>(pxbuf, g_pr, b_pr, H);
        gemm(pxbuf, H, WprT, bpr, nullptr, priors + (size_t)t * B * 2 * S,
             B, 2 * S, H);
    }

    (void)hipMemcpyAsync(hf, h_cur, (size_t)B * D * sizeof(float),
                         hipMemcpyDeviceToDevice, stream);
    (void)hipMemcpyAsync(zf, z_cur, (size_t)B * S * sizeof(float),
                         hipMemcpyDeviceToDevice, stream);
}
